// ConstraintModel_2491081031920
// MI455X (gfx1250) — compile-verified, matrix operand order checked by reading, not run on hardware
//
#include <hip/hip_runtime.h>
#include <hip/hip_bf16.h>
#include <stdint.h>
#include <stddef.h>

// ---------------------------------------------------------------------------
// Types / fragment helpers
// ---------------------------------------------------------------------------
typedef __bf16 bf16;
typedef __attribute__((ext_vector_type(16))) __bf16 v16bf;
typedef __attribute__((ext_vector_type(8)))  float  v8f;
typedef __attribute__((ext_vector_type(4)))  unsigned int u32x4;

union V16U { v16bf v; u32x4 q[2]; };

// Load a 16-element bf16 fragment (two contiguous 16B chunks) for WMMA A/B.
// Caller passes base = rowptr + half*8 (+ kb). Matches CDNA5 16-bit 16x32
// A layout: lanes 0-15 rows M, VGPR0-3 hold K=half*8+0..7, VGPR4-7 K=+16.
__device__ __forceinline__ v16bf load_frag(const bf16* p) {
  V16U u;
  u.q[0] = *reinterpret_cast<const u32x4*>(p);
  u.q[1] = *reinterpret_cast<const u32x4*>(p + 16);
  return u.v;
}

__device__ __forceinline__ float sigmoid_(float x) { return 1.f / (1.f + __expf(-x)); }
__device__ __forceinline__ float tanh_(float x)    { return 2.f / (1.f + __expf(-2.f * x)) - 1.f; }

#define WMMA_BF16(A, Bm, C) \
  __builtin_amdgcn_wmma_f32_16x16x32_bf16(false, (A), false, (Bm), (short)0, (C), false, false)

// ---------------------------------------------------------------------------
// Fused LSTM timestep: gates = [s0|s1|s2] @ Wcat^T + bias; cell update; h out.
// Wave computes one (16 batch x 16 hidden) tile => 4 gate tiles (i,f,g,o).
// grid = (32 hidden-blocks, 4), block = 128 (4 waves => 16 batch-blocks).
// ---------------------------------------------------------------------------
__global__ __launch_bounds__(128) void lstm_step(
    const bf16* __restrict__ s0, int l0, int rs0,
    const bf16* __restrict__ s1, int l1, int rs1,
    const bf16* __restrict__ s2, int l2, int rs2,   // h_{t-1}; null at t==0
    const bf16* __restrict__ Wcat, int Kpad,        // (2048 x Kpad) bf16
    const float* __restrict__ bias,                 // 2048 f32 (bih+bhh)
    float* __restrict__ Cst, int t0,                // c state (256 x 512) f32
    bf16* __restrict__ Hout)                        // h_t (256 x 512) bf16
{
  const int H = 512;
  const int wave = threadIdx.x >> 5;
  const int lane = threadIdx.x & 31;
  const int half = lane >> 4;
  const int l16  = lane & 15;
  const int nblk = blockIdx.x;               // hidden tile 0..31
  const int mblk = blockIdx.y * 4 + wave;    // batch tile 0..15
  const int arow = mblk * 16 + l16;          // batch row (A operand)
  const int wrow = nblk * 16 + l16;          // weight row within a gate (B op)

  v8f acc0 = {}, acc1 = {}, acc2 = {}, acc3 = {};

  const bf16* segs[3] = { s0, s1, s2 };
  const int   lens[3] = { l0, l1, l2 };
  const int   rss[3]  = { rs0, rs1, rs2 };

  int koff = 0;
  #pragma unroll
  for (int s = 0; s < 3; ++s) {
    const bf16* sp = segs[s];
    const int len = lens[s];
    if (sp != nullptr) {
      const bf16* abase = sp + (size_t)arow * rss[s] + half * 8;
      const bf16* wbase = Wcat + (size_t)wrow * Kpad + koff + half * 8;
      for (int kb = 0; kb < len; kb += 32) {
        v16bf a  = load_frag(abase + kb);
        v16bf b0 = load_frag(wbase + kb);
        v16bf b1 = load_frag(wbase + kb + (size_t)H * Kpad);
        v16bf b2 = load_frag(wbase + kb + (size_t)2 * H * Kpad);
        v16bf b3 = load_frag(wbase + kb + (size_t)3 * H * Kpad);
        acc0 = WMMA_BF16(a, b0, acc0);
        acc1 = WMMA_BF16(a, b1, acc1);
        acc2 = WMMA_BF16(a, b2, acc2);
        acc3 = WMMA_BF16(a, b3, acc3);
      }
    }
    koff += len;
  }

  const int j = nblk * 16 + l16;             // hidden index (C/D column)
  const float bI = bias[j];
  const float bF = bias[H + j];
  const float bG = bias[2 * H + j];
  const float bO = bias[3 * H + j];

  #pragma unroll
  for (int v = 0; v < 8; ++v) {
    const int brow = mblk * 16 + v + 8 * half;   // C/D row layout
    const float iv = sigmoid_(acc0[v] + bI);
    const float fv = sigmoid_(acc1[v] + bF);
    const float gv = tanh_(acc2[v] + bG);
    const float ov = sigmoid_(acc3[v] + bO);
    const size_t ci = (size_t)brow * H + j;
    const float cold = t0 ? 0.f : Cst[ci];
    const float cn = fv * cold + iv * gv;
    Cst[ci] = cn;
    Hout[ci] = (bf16)(ov * tanh_(cn));
  }
}

// ---------------------------------------------------------------------------
// Dense layers: out = act(A(MxK) @ W(NxK)^T + bias)
// ---------------------------------------------------------------------------
__global__ __launch_bounds__(128) void k_gemm_relu_bf16(
    const bf16* __restrict__ A, const bf16* __restrict__ W,
    const float* __restrict__ bias, bf16* __restrict__ out, int K, int N)
{
  const int wave = threadIdx.x >> 5, lane = threadIdx.x & 31;
  const int half = lane >> 4, l16 = lane & 15;
  const int nblk = blockIdx.x;
  const int mblk = blockIdx.y * 4 + wave;
  v8f acc = {};
  const bf16* ab = A + ((size_t)mblk * 16 + l16) * K + half * 8;
  const bf16* wb = W + ((size_t)nblk * 16 + l16) * K + half * 8;
  for (int kb = 0; kb < K; kb += 32) {
    v16bf a = load_frag(ab + kb);
    v16bf b = load_frag(wb + kb);
    acc = WMMA_BF16(a, b, acc);
  }
  const int j = nblk * 16 + l16;
  const float bj = bias[j];
  #pragma unroll
  for (int v = 0; v < 8; ++v) {
    const int r = mblk * 16 + v + 8 * half;
    const float x = acc[v] + bj;
    out[(size_t)r * N + j] = (bf16)(x > 0.f ? x : 0.f);
  }
}

__global__ __launch_bounds__(128) void k_gemm_f32out(
    const bf16* __restrict__ A, const bf16* __restrict__ W,
    const float* __restrict__ bias, float* __restrict__ out, int K, int N)
{
  const int wave = threadIdx.x >> 5, lane = threadIdx.x & 31;
  const int half = lane >> 4, l16 = lane & 15;
  const int nblk = blockIdx.x;
  const int mblk = blockIdx.y * 4 + wave;
  v8f acc = {};
  const bf16* ab = A + ((size_t)mblk * 16 + l16) * K + half * 8;
  const bf16* wb = W + ((size_t)nblk * 16 + l16) * K + half * 8;
  for (int kb = 0; kb < K; kb += 32) {
    v16bf a = load_frag(ab + kb);
    v16bf b = load_frag(wb + kb);
    acc = WMMA_BF16(a, b, acc);
  }
  const int j = nblk * 16 + l16;
  const float bj = bias[j];
  #pragma unroll
  for (int v = 0; v < 8; ++v) {
    const int r = mblk * 16 + v + 8 * half;
    out[(size_t)r * N + j] = acc[v] + bj;
  }
}

// ---------------------------------------------------------------------------
// Packing / conversion kernels (run once per call, fully parallel)
// ---------------------------------------------------------------------------
__global__ void k_build_X0(const float* __restrict__ src, bf16* __restrict__ dst,
                           int T, int B) {
  // dst (T,B,160) bf16, time-reversed seq_constraints (T,B,129), zero-padded K
  long idx = (long)blockIdx.x * blockDim.x + threadIdx.x;
  long total = (long)T * B * 160;
  if (idx >= total) return;
  int k = (int)(idx % 160);
  long r = idx / 160;
  int b = (int)(r % B);
  int t = (int)(r / B);
  float v = (k < 129) ? src[((long)(T - 1 - t) * B + b) * 129 + k] : 0.f;
  dst[idx] = (bf16)v;
}

__global__ void k_build_Xg(const float* __restrict__ seq, bf16* __restrict__ dst,
                           int T, int B, int F) {
  // dst (T,B,F) bf16 = seq shifted by one step (t=0 -> zeros)
  long idx = (long)blockIdx.x * blockDim.x + threadIdx.x;
  long total = (long)T * B * F;
  if (idx >= total) return;
  long r = idx / F;
  int t = (int)(r / B);
  float v = (t == 0) ? 0.f : seq[idx - (long)B * F];
  dst[idx] = (bf16)v;
}

__global__ void k_build_wcat(const float* __restrict__ Wih, int inr, int inp,
                             const float* __restrict__ Whh,
                             bf16* __restrict__ dst, int Kpad) {
  // dst (2048 x Kpad): [Wih (zero-padded to inp) | Whh(512)] per row
  long idx = (long)blockIdx.x * blockDim.x + threadIdx.x;
  long total = 2048L * Kpad;
  if (idx >= total) return;
  int k = (int)(idx % Kpad);
  long row = idx / Kpad;
  float v;
  if (k < inp) v = (k < inr) ? Wih[row * inr + k] : 0.f;
  else         v = Whh[row * 512 + (k - inp)];
  dst[idx] = (bf16)v;
}

__global__ void k_bias_sum(const float* __restrict__ a, const float* __restrict__ b,
                           float* __restrict__ o, int n) {
  int i = blockIdx.x * blockDim.x + threadIdx.x;
  if (i < n) o[i] = a[i] + b[i];
}

__global__ void k_cvt_bf16(const float* __restrict__ in, bf16* __restrict__ out, long n) {
  long i = (long)blockIdx.x * blockDim.x + threadIdx.x;
  if (i < n) out[i] = (bf16)in[i];
}

// ---------------------------------------------------------------------------
// Host driver
// ---------------------------------------------------------------------------
static inline long cdiv(long a, long b) { return (a + b - 1) / b; }

extern "C" void kernel_launch(void* const* d_in, const int* in_sizes, int n_in,
                              void* d_out, int out_size, void* d_ws, size_t ws_size,
                              hipStream_t stream) {
  (void)in_sizes; (void)n_in; (void)out_size;
  const int T = 1024, B = 256, F = 128, H = 512, Lh = 256;
  const long M = (long)T * B;   // 262144 rows for dense layers

  const float* seq    = (const float*)d_in[0];
  const float* seqc   = (const float*)d_in[1];
  const float* c0_Wih = (const float*)d_in[2];
  const float* c0_Whh = (const float*)d_in[3];
  const float* c0_bih = (const float*)d_in[4];
  const float* c0_bhh = (const float*)d_in[5];
  const float* c1_Wih = (const float*)d_in[6];
  const float* c1_Whh = (const float*)d_in[7];
  const float* c1_bih = (const float*)d_in[8];
  const float* c1_bhh = (const float*)d_in[9];
  const float* g0_Wih = (const float*)d_in[10];
  const float* g0_Whh = (const float*)d_in[11];
  const float* g0_bih = (const float*)d_in[12];
  const float* g0_bhh = (const float*)d_in[13];
  const float* g1_Wih = (const float*)d_in[14];
  const float* g1_Whh = (const float*)d_in[15];
  const float* g1_bih = (const float*)d_in[16];
  const float* g1_bhh = (const float*)d_in[17];
  const float* lin1_W = (const float*)d_in[18];
  const float* lin1_b = (const float*)d_in[19];
  const float* lin2_W = (const float*)d_in[20];
  const float* lin2_b = (const float*)d_in[21];

  // ---- workspace layout (256B aligned regions) ----
  size_t off = 0;
  auto take = [&](size_t bytes) { size_t o = off; off = (off + bytes + 255) & ~(size_t)255; return o; };
  const size_t oX0  = take((size_t)T * B * 160 * 2);     // reversed constraints, K-padded
  const size_t oXg  = take((size_t)T * B * F   * 2);     // shifted seq
  const size_t oHA  = take((size_t)T * B * H   * 2);     // ping
  const size_t oHB  = take((size_t)T * B * H   * 2);     // pong
  const size_t oH1  = take((size_t)M * Lh * 2);          // relu(lin1) output
  const size_t oWc0 = take((size_t)2048 * 672  * 2);
  const size_t oWc1 = take((size_t)2048 * 1024 * 2);
  const size_t oWg0 = take((size_t)2048 * 1152 * 2);
  const size_t oWg1 = take((size_t)2048 * 1024 * 2);
  const size_t oWl1 = take((size_t)Lh * H * 2);
  const size_t oWl2 = take((size_t)F * Lh * 2);
  const size_t oBs  = take((size_t)4 * 2048 * 4);
  const size_t oCst = take((size_t)B * H * 4);
  if (ws_size < off || d_ws == nullptr) return;          // scratch too small: bail

  char* ws = (char*)d_ws;
  bf16*  X0  = (bf16*)(ws + oX0);
  bf16*  Xg  = (bf16*)(ws + oXg);
  bf16*  HA  = (bf16*)(ws + oHA);
  bf16*  HB  = (bf16*)(ws + oHB);
  bf16*  H1  = (bf16*)(ws + oH1);
  bf16*  Wc0 = (bf16*)(ws + oWc0);
  bf16*  Wc1 = (bf16*)(ws + oWc1);
  bf16*  Wg0 = (bf16*)(ws + oWg0);
  bf16*  Wg1 = (bf16*)(ws + oWg1);
  bf16*  Wl1 = (bf16*)(ws + oWl1);
  bf16*  Wl2 = (bf16*)(ws + oWl2);
  float* Bs  = (float*)(ws + oBs);                        // 4 x 2048
  float* Cst = (float*)(ws + oCst);

  const int TPB = 256;
  // ---- one-time packing ----
  k_build_X0<<<dim3((unsigned)cdiv((long)T * B * 160, TPB)), TPB, 0, stream>>>(seqc, X0, T, B);
  k_build_Xg<<<dim3((unsigned)cdiv((long)T * B * F, TPB)), TPB, 0, stream>>>(seq, Xg, T, B, F);
  k_build_wcat<<<dim3((unsigned)cdiv(2048L * 672, TPB)), TPB, 0, stream>>>(c0_Wih, 129, 160, c0_Whh, Wc0, 672);
  k_build_wcat<<<dim3((unsigned)cdiv(2048L * 1024, TPB)), TPB, 0, stream>>>(c1_Wih, 512, 512, c1_Whh, Wc1, 1024);
  k_build_wcat<<<dim3((unsigned)cdiv(2048L * 1152, TPB)), TPB, 0, stream>>>(g0_Wih, 640, 640, g0_Whh, Wg0, 1152);
  k_build_wcat<<<dim3((unsigned)cdiv(2048L * 1024, TPB)), TPB, 0, stream>>>(g1_Wih, 512, 512, g1_Whh, Wg1, 1024);
  k_bias_sum<<<8, TPB, 0, stream>>>(c0_bih, c0_bhh, Bs + 0 * 2048, 2048);
  k_bias_sum<<<8, TPB, 0, stream>>>(c1_bih, c1_bhh, Bs + 1 * 2048, 2048);
  k_bias_sum<<<8, TPB, 0, stream>>>(g0_bih, g0_bhh, Bs + 2 * 2048, 2048);
  k_bias_sum<<<8, TPB, 0, stream>>>(g1_bih, g1_bhh, Bs + 3 * 2048, 2048);
  k_cvt_bf16<<<dim3((unsigned)cdiv((long)Lh * H, TPB)), TPB, 0, stream>>>(lin1_W, Wl1, (long)Lh * H);
  k_cvt_bf16<<<dim3((unsigned)cdiv((long)F * Lh, TPB)), TPB, 0, stream>>>(lin2_W, Wl2, (long)F * Lh);

  const dim3 sGrid(32, 4), sBlk(128);
  const size_t hstep = (size_t)B * H;      // elements per timestep of an H-sequence

  // ---- layer c0: X0 -> HA (forward scan over reversed input) ----
  for (int t = 0; t < T; ++t) {
    lstm_step<<<sGrid, sBlk, 0, stream>>>(
        X0 + (size_t)t * B * 160, 160, 160,
        nullptr, 0, 0,
        t ? HA + (size_t)(t - 1) * hstep : nullptr, 512, 512,
        Wc0, 672, Bs + 0 * 2048, Cst, t == 0, HA + (size_t)t * hstep);
  }
  // ---- layer c1: HA -> HB ----
  for (int t = 0; t < T; ++t) {
    lstm_step<<<sGrid, sBlk, 0, stream>>>(
        HA + (size_t)t * hstep, 512, 512,
        nullptr, 0, 0,
        t ? HB + (size_t)(t - 1) * hstep : nullptr, 512, 512,
        Wc1, 1024, Bs + 1 * 2048, Cst, t == 0, HB + (size_t)t * hstep);
  }
  // ---- layer g0: [Xg | reverse(HB)] -> HA ----
  for (int t = 0; t < T; ++t) {
    lstm_step<<<sGrid, sBlk, 0, stream>>>(
        Xg + (size_t)t * B * F, 128, 128,
        HB + (size_t)(T - 1 - t) * hstep, 512, 512,   // out_constraints[t]
        t ? HA + (size_t)(t - 1) * hstep : nullptr, 512, 512,
        Wg0, 1152, Bs + 2 * 2048, Cst, t == 0, HA + (size_t)t * hstep);
  }
  // ---- layer g1: HA -> HB ----
  for (int t = 0; t < T; ++t) {
    lstm_step<<<sGrid, sBlk, 0, stream>>>(
        HA + (size_t)t * hstep, 512, 512,
        nullptr, 0, 0,
        t ? HB + (size_t)(t - 1) * hstep : nullptr, 512, 512,
        Wg1, 1024, Bs + 3 * 2048, Cst, t == 0, HB + (size_t)t * hstep);
  }

  // ---- dense head: h1 = relu(HB @ lin1_W^T + b); preds = h1 @ lin2_W^T + b ----
  k_gemm_relu_bf16<<<dim3(Lh / 16, (unsigned)(M / 64)), 128, 0, stream>>>(
      HB, Wl1, lin1_b, H1, H, Lh);
  k_gemm_f32out<<<dim3(F / 16, (unsigned)(M / 64)), 128, 0, stream>>>(
      H1, Wl2, lin2_b, (float*)d_out, Lh, F);
}
